// CustomCrossAttention_49220325212512
// MI455X (gfx1250) — matrix operand order repaired and checked by reading, not compiled, not run on hardware
//
#include <hip/hip_runtime.h>
#include <hip/hip_bf16.h>

#define B_   4
#define C_   64
#define N_   4096      // H*W = 64*64
#define DQ_  64
#define FP_  128
#define SCALE_ 0.125f
#define LOG2E_ 1.44269504088896f

#define JSTEP   32
#define KSTRIDE 72     // padded LDS row stride (halves) for K tile, 16B aligned
#define VSTRIDE 40     // padded LDS row stride (halves) for V tile, 16B aligned

typedef __attribute__((ext_vector_type(16))) __bf16 v16bf;
typedef __attribute__((ext_vector_type(8)))  float  v8f;
typedef __attribute__((ext_vector_type(4)))  int    v4i_;

union FragBF { v16bf v; uint4 q[2]; unsigned int u[8]; };
union AccF   { v8f v; float f[8]; };

#if defined(__has_builtin)
#if __has_builtin(__builtin_amdgcn_global_load_async_to_lds_b128)
#define HAVE_ASYNC_LDS 1
#endif
#endif
#ifndef HAVE_ASYNC_LDS
#define HAVE_ASYNC_LDS 0
#endif

__device__ __forceinline__ void async_b128(const unsigned short* gsrc,
                                           unsigned short* ldst) {
#if HAVE_ASYNC_LDS
    // Builtin signature (from clang diagnostic): (v4i* [AS1], v4i* [AS3],
    // imm offset, imm cpol). Generic->device/shared AS conversion is implicit
    // in HIP device code once the pointee type matches.
    __builtin_amdgcn_global_load_async_to_lds_b128(
        (v4i_*)gsrc, (v4i_*)ldst, 0, 0);
#else
    *(uint4*)ldst = *(const uint4*)gsrc;   // fallback: load + ds_store staging
#endif
}

__device__ __forceinline__ void wait_async_lds() {
#if HAVE_ASYNC_LDS
#if __has_builtin(__builtin_amdgcn_s_wait_asynccnt)
    __builtin_amdgcn_s_wait_asynccnt(0);
#else
    asm volatile("s_wait_asynccnt 0x0" ::: "memory");
#endif
#endif
}

__device__ __forceinline__ unsigned short f2bf(float f) {
    unsigned int u = __float_as_uint(f);
    unsigned int r = u + 0x7FFFu + ((u >> 16) & 1u);   // round-to-nearest-even
    return (unsigned short)(r >> 16);
}
__device__ __forceinline__ unsigned int pack2bf(float lo, float hi) {
    return (unsigned int)f2bf(lo) | ((unsigned int)f2bf(hi) << 16);
}

// ---------------------------------------------------------------------------
// Kernel 0: fp_proj[b][d] = bfp[d] + dot(Wfp[d,:], fingerprint[b,:])
// ---------------------------------------------------------------------------
__global__ void fpproj_kernel(const float* __restrict__ fingerprint,
                              const float* __restrict__ Wfp,
                              const float* __restrict__ bfp,
                              float* __restrict__ fp_proj) {
    int t = blockIdx.x * blockDim.x + threadIdx.x;
    if (t >= B_ * DQ_) return;
    int b = t >> 6, d = t & 63;
    float acc = bfp[d];
    const float* fp = fingerprint + b * FP_;
    const float* w  = Wfp + d * FP_;
    #pragma unroll 8
    for (int i = 0; i < FP_; ++i) acc = fmaf(w[i], fp[i], acc);
    fp_proj[t] = acc;
}

// ---------------------------------------------------------------------------
// Kernel 1: Q/K/V 1x1-conv projections + fingerprint gate map.
//   x: [B][C][N] f32.  Outputs: qb,kb bf16 [B][N][64]; vb bf16 [B][64][N];
//   amap f32 [B][N] = sigmoid(<q[:,n], fp_proj[b,:]>)
// ---------------------------------------------------------------------------
__global__ __launch_bounds__(256)
void proj_kernel(const float* __restrict__ x,
                 const float* __restrict__ Wq, const float* __restrict__ bq,
                 const float* __restrict__ Wk, const float* __restrict__ bk,
                 const float* __restrict__ Wv, const float* __restrict__ bv,
                 const float* __restrict__ fp_proj,
                 unsigned short* __restrict__ qb,
                 unsigned short* __restrict__ kb,
                 unsigned short* __restrict__ vb,
                 float* __restrict__ amap) {
    const int b = blockIdx.y;
    const int n = blockIdx.x * blockDim.x + threadIdx.x;
    const float* xB = x + (size_t)b * C_ * N_ + n;
    float xc[C_];
    #pragma unroll
    for (int c = 0; c < C_; ++c) xc[c] = xB[(size_t)c * N_];

    float am = 0.0f;
    for (int d = 0; d < DQ_; ++d) {
        float aq = bq[d], ak = bk[d], av = bv[d];
        const float* wq = Wq + d * C_;
        const float* wk = Wk + d * C_;
        const float* wv = Wv + d * C_;
        #pragma unroll
        for (int c = 0; c < C_; ++c) {
            aq = fmaf(wq[c], xc[c], aq);
            ak = fmaf(wk[c], xc[c], ak);
            av = fmaf(wv[c], xc[c], av);
        }
        qb[((size_t)b * N_ + n) * DQ_ + d] = f2bf(aq);
        kb[((size_t)b * N_ + n) * DQ_ + d] = f2bf(ak);
        vb[((size_t)b * C_ + d) * N_ + n]  = f2bf(av);
        am = fmaf(aq, fp_proj[b * DQ_ + d], am);
    }
    amap[b * N_ + n] = 1.0f / (1.0f + exp2f(-am * LOG2E_));
}

// ---------------------------------------------------------------------------
// Kernel 2: flash attention. Workgroup = 4 waves, each wave owns 16 query
// columns; the 4 waves share double-buffered K/V tiles staged in LDS via
// gfx1250 async global->LDS copies (ASYNCcnt + s_wait_asynccnt), overlapping
// the next tile's DMA with the current tile's WMMA + online softmax.
//   attnout: f32 [B][64][N]
// ---------------------------------------------------------------------------
__global__ __launch_bounds__(128)
void flash_attn_kernel(const unsigned short* __restrict__ qbuf,
                       const unsigned short* __restrict__ kbuf,
                       const unsigned short* __restrict__ vbuf,
                       float* __restrict__ attnout) {
    __shared__ unsigned short ldsK[2][JSTEP * KSTRIDE];  // [j][d], padded rows
    __shared__ unsigned short ldsV[2][C_ * VSTRIDE];     // [c][j], padded rows

    const int tid  = threadIdx.x;
    const int b    = blockIdx.y;
    const int lane = tid & 31;
    const int wave = tid >> 5;
    const int i0   = blockIdx.x * 64 + wave * 16;
    const int half = lane >> 4;      // 0: lanes 0-15, 1: lanes 16-31
    const int l16  = lane & 15;

    const unsigned short* qB = qbuf + (size_t)b * N_ * DQ_;
    const unsigned short* kB = kbuf + (size_t)b * N_ * DQ_;
    const unsigned short* vB = vbuf + (size_t)b * C_ * N_;

    // Cooperative async fetch of one 32-key step's K tile (32x64) + V tile
    // (64x32) into LDS buffer `buf`: 512 x 16B chunks, 4 per thread.
    auto fetch_tiles = [&](int j0, int buf) {
        #pragma unroll
        for (int c = tid; c < 256; c += 128) {           // K: 8 chunks/row
            const int row = c >> 3, blk = c & 7;
            async_b128(kB + (size_t)(j0 + row) * DQ_ + blk * 8,
                       &ldsK[buf][row * KSTRIDE + blk * 8]);
        }
        #pragma unroll
        for (int c = tid; c < 256; c += 128) {           // V: 4 chunks/row
            const int row = c >> 2, blk = c & 3;
            async_b128(vB + (size_t)row * N_ + j0 + blk * 8,
                       &ldsV[buf][row * VSTRIDE + blk * 8]);
        }
    };

    // Persistent Q B-fragments: columns i0..i0+15, d in [0,32) and [32,64).
    // B layout: lane<16 holds K=0..7,16..23; lane>=16 holds K=8..15,24..31.
    FragBF qf0, qf1;
    {
        const unsigned short* p = qB + (size_t)(i0 + l16) * DQ_ + half * 8;
        qf0.q[0] = *(const uint4*)(p + 0);
        qf0.q[1] = *(const uint4*)(p + 16);
        qf1.q[0] = *(const uint4*)(p + 32);
        qf1.q[1] = *(const uint4*)(p + 48);
    }

    AccF o[4];
    #pragma unroll
    for (int t = 0; t < 4; ++t)
        #pragma unroll
        for (int r = 0; r < 8; ++r) o[t].f[r] = 0.0f;

    float m = -1e30f, l = 0.0f;
    const float qk_scale = SCALE_ * LOG2E_;

    fetch_tiles(0, 0);
    wait_async_lds();
    __syncthreads();

    for (int j0 = 0; j0 < N_; j0 += JSTEP) {
        const int buf = (j0 >> 5) & 1;
        if (j0 + JSTEP < N_) fetch_tiles(j0 + JSTEP, buf ^ 1);

        // A-frags of K^T from LDS: rows j, K dim = d. Per-lane: halves 0..7 =
        // d = koff..koff+7, halves 8..15 = d = 16+koff.. (koff = half*8).
        FragBF ka0, ka1, kc0, kc1;
        {
            const unsigned short* p0 = &ldsK[buf][l16 * KSTRIDE + half * 8];
            const unsigned short* p1 = &ldsK[buf][(16 + l16) * KSTRIDE + half * 8];
            ka0.q[0] = *(const uint4*)(p0);      ka0.q[1] = *(const uint4*)(p0 + 16);
            ka1.q[0] = *(const uint4*)(p0 + 32); ka1.q[1] = *(const uint4*)(p0 + 48);
            kc0.q[0] = *(const uint4*)(p1);      kc0.q[1] = *(const uint4*)(p1 + 16);
            kc1.q[0] = *(const uint4*)(p1 + 32); kc1.q[1] = *(const uint4*)(p1 + 48);
        }

        // S^T tiles (M = j, N = i), accumulate over d via chained WMMA.
        AccF s0, s1;
        #pragma unroll
        for (int r = 0; r < 8; ++r) { s0.f[r] = 0.0f; s1.f[r] = 0.0f; }
        s0.v = __builtin_amdgcn_wmma_f32_16x16x32_bf16(false, ka0.v, false, qf0.v,
                                                       (short)0, s0.v, false, false);
        s0.v = __builtin_amdgcn_wmma_f32_16x16x32_bf16(false, ka1.v, false, qf1.v,
                                                       (short)0, s0.v, false, false);
        s1.v = __builtin_amdgcn_wmma_f32_16x16x32_bf16(false, kc0.v, false, qf0.v,
                                                       (short)0, s1.v, false, false);
        s1.v = __builtin_amdgcn_wmma_f32_16x16x32_bf16(false, kc1.v, false, qf1.v,
                                                       (short)0, s1.v, false, false);

        // Online softmax over j (rows of S^T): each lane holds 16 of the 32 j's
        // for its column i; the other 16 live in the xor-16 partner lane.
        float tmax = -1e30f;
        #pragma unroll
        for (int r = 0; r < 8; ++r) {
            s0.f[r] *= qk_scale; s1.f[r] *= qk_scale;
            tmax = fmaxf(tmax, fmaxf(s0.f[r], s1.f[r]));
        }
        tmax = fmaxf(tmax, __shfl_xor(tmax, 16, 32));
        const float mnew  = fmaxf(m, tmax);
        const float alpha = exp2f(m - mnew);
        m = mnew;

        float e0[8], e1[8], lsum = 0.0f;
        #pragma unroll
        for (int r = 0; r < 8; ++r) {
            e0[r] = exp2f(s0.f[r] - m);
            e1[r] = exp2f(s1.f[r] - m);
            lsum += e0[r] + e1[r];
        }
        lsum += __shfl_xor(lsum, 16, 32);
        l = l * alpha + lsum;

        #pragma unroll
        for (int t = 0; t < 4; ++t)
            #pragma unroll
            for (int r = 0; r < 8; ++r) o[t].f[r] *= alpha;

        // Build P B-fragment (K = 32 j's, columns i). Lane<16 needs j 0..15:
        // own tile0 rows (j 0..7) + partner tile0 rows (j 8..15). Lane>=16
        // needs j 16..31: partner tile1 (16..23) + own tile1 (24..31).
        unsigned int t0p[4], t1p[4], x0[4], x1[4];
        #pragma unroll
        for (int v2 = 0; v2 < 4; ++v2) {
            t0p[v2] = pack2bf(e0[2 * v2], e0[2 * v2 + 1]);
            t1p[v2] = pack2bf(e1[2 * v2], e1[2 * v2 + 1]);
        }
        #pragma unroll
        for (int v2 = 0; v2 < 4; ++v2) {
            x0[v2] = __shfl_xor(t0p[v2], 16, 32);
            x1[v2] = __shfl_xor(t1p[v2], 16, 32);
        }
        FragBF pB;
        #pragma unroll
        for (int v2 = 0; v2 < 4; ++v2) {
            pB.u[v2]     = half ? x1[v2]  : t0p[v2];
            pB.u[4 + v2] = half ? t1p[v2] : x0[v2];
        }

        // O += V * P  (A = V rows c from LDS, K = j; 4 c-tiles of 16)
        #pragma unroll
        for (int ct = 0; ct < 4; ++ct) {
            FragBF va;
            const unsigned short* p = &ldsV[buf][(ct * 16 + l16) * VSTRIDE + half * 16];
            va.q[0] = *(const uint4*)p;
            va.q[1] = *(const uint4*)(p + 8);
            o[ct].v = __builtin_amdgcn_wmma_f32_16x16x32_bf16(false, va.v, false, pB.v,
                                                              (short)0, o[ct].v, false, false);
        }

        wait_async_lds();   // next tile landed; also fences before buffer reuse
        __syncthreads();
    }

    const float rinv = 1.0f / l;
    #pragma unroll
    for (int ct = 0; ct < 4; ++ct)
        #pragma unroll
        for (int r = 0; r < 8; ++r) {
            const int c = ct * 16 + r + half * 8;
            attnout[((size_t)b * C_ + c) * N_ + i0 + l16] = o[ct].f[r] * rinv;
        }
}

// ---------------------------------------------------------------------------
// Kernel 3: out = Wo @ (attnout * gate + x) + bo
// ---------------------------------------------------------------------------
__global__ __launch_bounds__(256)
void out_kernel(const float* __restrict__ x,
                const float* __restrict__ attnout,
                const float* __restrict__ amap,
                const float* __restrict__ Wo, const float* __restrict__ bo,
                float* __restrict__ out) {
    const int b = blockIdx.y;
    const int n = blockIdx.x * blockDim.x + threadIdx.x;
    const float g = amap[b * N_ + n];
    float t[C_];
    #pragma unroll
    for (int c = 0; c < C_; ++c) {
        const size_t idx = ((size_t)b * C_ + c) * N_ + n;
        t[c] = fmaf(attnout[idx], g, x[idx]);
    }
    for (int oc = 0; oc < C_; ++oc) {
        float acc = bo[oc];
        const float* w = Wo + oc * C_;
        #pragma unroll
        for (int c = 0; c < C_; ++c) acc = fmaf(w[c], t[c], acc);
        out[((size_t)b * C_ + oc) * N_ + n] = acc;
    }
}

// ---------------------------------------------------------------------------
extern "C" void kernel_launch(void* const* d_in, const int* in_sizes, int n_in,
                              void* d_out, int out_size, void* d_ws, size_t ws_size,
                              hipStream_t stream) {
    const float* x           = (const float*)d_in[0];
    const float* fingerprint = (const float*)d_in[1];
    const float* Wq  = (const float*)d_in[2];
    const float* bq  = (const float*)d_in[3];
    const float* Wk  = (const float*)d_in[4];
    const float* bk  = (const float*)d_in[5];
    const float* Wv  = (const float*)d_in[6];
    const float* bv  = (const float*)d_in[7];
    const float* Wo  = (const float*)d_in[8];
    const float* bo  = (const float*)d_in[9];
    const float* Wfp = (const float*)d_in[10];
    const float* bfp = (const float*)d_in[11];
    float* out = (float*)d_out;

    // Workspace layout (16B-aligned sections): ~10.1 MB total.
    char* ws = (char*)d_ws;
    float* fp_proj = (float*)ws;                                   //  1 KB
    float* amap    = (float*)(ws + 1024);                          // 64 KB
    unsigned short* qb = (unsigned short*)(ws + 1024 + 65536);     // 2 MB
    unsigned short* kb = qb + (size_t)B_ * N_ * DQ_;               // 2 MB
    unsigned short* vb = kb + (size_t)B_ * N_ * DQ_;               // 2 MB
    float* attnout = (float*)(vb + (size_t)B_ * C_ * N_);          // 4 MB

    fpproj_kernel<<<1, 256, 0, stream>>>(fingerprint, Wfp, bfp, fp_proj);

    dim3 gp(N_ / 256, B_);
    proj_kernel<<<gp, 256, 0, stream>>>(x, Wq, bq, Wk, bk, Wv, bv, fp_proj,
                                        qb, kb, vb, amap);

    dim3 gf(N_ / 64, B_);
    flash_attn_kernel<<<gf, 128, 0, stream>>>(qb, kb, vb, attnout);

    out_kernel<<<gp, 256, 0, stream>>>(x, attnout, amap, Wo, bo, out);
}
